// OriFlow_64896955842658
// MI455X (gfx1250) — compile-verified, matrix operand order checked by reading, not run on hardware
//
#include <hip/hip_runtime.h>
#include <math.h>

// ---- problem constants (from reference) ----
#define BN    2
#define SN    1024
#define DIMN  256
#define DIN   512
#define NSN   16
#define DTRN  16
#define KCN   4
#define LN    4
#define CDN   512
#define BSN   (BN*SN)
#define XDTN  (DTRN + 2*NSN)   // 48 (logical)
#define XDTP  64               // padded row stride for xdt activation buffer
#define MT    4                // M-tiles (16 rows each) per wave in GEMM

typedef __attribute__((ext_vector_type(16))) __bf16 v16bf;
typedef __attribute__((ext_vector_type(8)))  __bf16 v8bf;
typedef __attribute__((ext_vector_type(8)))  float  v8f;

__device__ __forceinline__ float siluf(float x) { return x / (1.0f + __expf(-x)); }

// ============================================================================
// Weight repack: W[K][N] fp32 -> Wt[N][Kp] bf16, zero-padded rows k in [K,Kp).
// ============================================================================
__global__ void repack_weight_kernel(const float* __restrict__ W,
                                     __bf16* __restrict__ Wt,
                                     int K, int N, int Kp)
{
    int idx = blockIdx.x * blockDim.x + threadIdx.x;
    if (idx >= N * Kp) return;
    int k = idx % Kp;
    int n = idx / Kp;
    float v = (k < K) ? W[(size_t)k * N + n] : 0.0f;
    Wt[idx] = (__bf16)v;
}

__global__ void cast_bf16_kernel(const float* __restrict__ in,
                                 __bf16* __restrict__ out, int n)
{
    int idx = blockIdx.x * blockDim.x + threadIdx.x;
    if (idx < n) out[idx] = (__bf16)in[idx];
}

// ============================================================================
// WMMA GEMM: C = act(A[M,Kp](bf16, row stride lda) * Wt[N,Kp](bf16) + bias)
// A and Wt both bf16 in fragment-ready layout: per lane, per k-step, a fragment
// is two contiguous 16-byte runs -> pure global_load_b128, zero conversions.
// Each wave: 4 M-tiles x 1 N-tile (64x16 of C), shared B fragment, 4 WMMA/step.
// Optional bf16 shadow output Cb (row stride ldcb) for chained GEMMs.
// block = (32, 8).  grid = (ceil(N/16/8), M/64).  act: 0 none, 1 silu, 2 softplus
// ============================================================================
__global__ void wmma_gemm_kernel(const __bf16* __restrict__ A, int lda,
                                 const __bf16* __restrict__ Wt, int Kp,
                                 const float* __restrict__ bias,
                                 float* __restrict__ C, int ldc,
                                 __bf16* __restrict__ Cb, int ldcb,
                                 int N, int act)
{
    const int lane = threadIdx.x;        // 0..31
    const int half = lane >> 4;          // 0 or 1
    const int l15  = lane & 15;
    const int ntile = blockIdx.x * blockDim.y + threadIdx.y;
    if (ntile * 16 >= N) return;         // uniform per wave
    const int m0   = blockIdx.y * (16 * MT);
    const int bcol = ntile * 16 + l15;   // B column for this lane

    const __bf16* __restrict__ bptr = Wt + (size_t)bcol * Kp + half * 8;
    const __bf16* __restrict__ aptr = A + (size_t)(m0 + l15) * lda + half * 8;
    const size_t astride = (size_t)16 * lda;   // 16 rows between M-tiles

    v8f acc0 = {}, acc1 = {}, acc2 = {}, acc3 = {};

    for (int k0 = 0; k0 < Kp; k0 += 32) {
        __builtin_prefetch(aptr + k0 + 64, 0, 1);
        // ---- B fragment: two contiguous 8-bf16 runs (ISA 7.12.2 layout) ----
        v8bf b0 = *(const v8bf*)(bptr + k0);
        v8bf b1 = *(const v8bf*)(bptr + k0 + 16);
        v16bf bf = __builtin_shufflevector(b0, b1,
                       0,1,2,3,4,5,6,7,8,9,10,11,12,13,14,15);
        // ---- A fragments: 4 M-tiles, two b128 loads each, no cvt ----
#pragma unroll
        for (int t = 0; t < MT; ++t) {
            const __bf16* ap = aptr + (size_t)t * astride + k0;
            v8bf a0 = *(const v8bf*)(ap);
            v8bf a1 = *(const v8bf*)(ap + 16);
            v16bf af = __builtin_shufflevector(a0, a1,
                           0,1,2,3,4,5,6,7,8,9,10,11,12,13,14,15);
            if (t == 0)      acc0 = __builtin_amdgcn_wmma_f32_16x16x32_bf16(false, af, false, bf, (short)0, acc0, false, false);
            else if (t == 1) acc1 = __builtin_amdgcn_wmma_f32_16x16x32_bf16(false, af, false, bf, (short)0, acc1, false, false);
            else if (t == 2) acc2 = __builtin_amdgcn_wmma_f32_16x16x32_bf16(false, af, false, bf, (short)0, acc2, false, false);
            else             acc3 = __builtin_amdgcn_wmma_f32_16x16x32_bf16(false, af, false, bf, (short)0, acc3, false, false);
        }
    }

    const float bv = bias ? bias[bcol] : 0.0f;
    // f32 C/D layout: lane<16 -> n, m=m0+r ; lane>=16 -> m=m0+8+r
#pragma unroll
    for (int t = 0; t < MT; ++t) {
        v8f acc = (t == 0) ? acc0 : (t == 1) ? acc1 : (t == 2) ? acc2 : acc3;
#pragma unroll
        for (int r = 0; r < 8; ++r) {
            int m = m0 + t * 16 + r + half * 8;
            float v = acc[r] + bv;
            if (act == 1)      v = siluf(v);
            else if (act == 2) v = (v > 20.0f) ? v : log1pf(__expf(v));
            C[(size_t)m * ldc + bcol] = v;
            if (Cb) Cb[(size_t)m * ldcb + bcol] = (__bf16)v;
        }
    }
}

// ============================================================================
// Time embedding: sin/cos features -> gelu MLP -> t_out[B, CD]
// ============================================================================
__global__ void time_embed_kernel(const float* __restrict__ times,
                                  const float* __restrict__ w1, const float* __restrict__ b1,
                                  const float* __restrict__ w2, const float* __restrict__ b2,
                                  float* __restrict__ t_out)
{
    const int b   = blockIdx.x;
    const int tid = threadIdx.x;            // 256 threads
    __shared__ float tv[CDN];
    __shared__ float h1[CDN];
    const int half = CDN / 2;               // 256
    float tm = times[b];
    {
        int i = tid;
        float f = __expf((float)i * (-logf(10000.0f) / (float)(half - 1)));
        float e = tm * f;
        tv[i]        = sinf(e);
        tv[half + i] = cosf(e);
    }
    __syncthreads();
    for (int o = tid; o < CDN; o += 256) {
        float acc = b1[o];
        for (int k = 0; k < CDN; ++k) acc += tv[k] * w1[(size_t)k * CDN + o];
        h1[o] = 0.5f * acc * (1.0f + erff(acc * 0.70710678f));   // exact gelu
    }
    __syncthreads();
    for (int o = tid; o < CDN; o += 256) {
        float acc = b2[o];
        for (int k = 0; k < CDN; ++k) acc += h1[k] * w2[(size_t)k * CDN + o];
        t_out[(size_t)b * CDN + o] = acc;
    }
}

// ============================================================================
// Encoder per-layer modulation: out[i][b][:] = silu(t[b]) @ mlp_w[i] + mlp_b[i]
// ============================================================================
__global__ void enc_cmod_kernel(const float* __restrict__ t,
                                const float* __restrict__ mlp_w,  // L x CD x 2DIM
                                const float* __restrict__ mlp_b,  // L x 2DIM
                                float* __restrict__ out)          // L x B x 2DIM
{
    const int i = blockIdx.x, b = blockIdx.y;
    const int tid = threadIdx.x;            // 256
    __shared__ float st[CDN];
    for (int k = tid; k < CDN; k += 256) st[k] = siluf(t[(size_t)b * CDN + k]);
    __syncthreads();
    const float* w = mlp_w + (size_t)i * CDN * (2 * DIMN);
    for (int o = tid; o < 2 * DIMN; o += 256) {
        float acc = mlp_b[(size_t)i * 2 * DIMN + o];
        for (int k = 0; k < CDN; ++k) acc += st[k] * w[(size_t)k * (2 * DIMN) + o];
        out[((size_t)i * BN + b) * (2 * DIMN) + o] = acc;
    }
}

// ============================================================================
// Initial circular conv: noised (B,S,1) -> x (B,S,DIM), width-3 circular
// ============================================================================
__global__ void init_conv_kernel(const float* __restrict__ noised,
                                 const float* __restrict__ w,   // DIM x 1 x 3
                                 const float* __restrict__ bias,
                                 float* __restrict__ x)
{
    int idx = blockIdx.x * blockDim.x + threadIdx.x;
    if (idx >= BSN * DIMN) return;
    int d = idx & (DIMN - 1);
    int row = idx >> 8;
    int s = row & (SN - 1);
    int b = row / SN;
    int sm = (s + SN - 1) & (SN - 1), sp = (s + 1) & (SN - 1);
    float v = w[d * 3 + 0] * noised[(size_t)b * SN + sm]
            + w[d * 3 + 1] * noised[(size_t)b * SN + s]
            + w[d * 3 + 2] * noised[(size_t)b * SN + sp]
            + bias[d];
    x[idx] = v;
}

// ============================================================================
// Final circular conv: x (B,S,DIM) -> out (B,S)
// ============================================================================
__global__ void final_conv_kernel(const float* __restrict__ x,
                                  const float* __restrict__ w,   // 1 x DIM x 3
                                  const float* __restrict__ bias,
                                  float* __restrict__ out)
{
    int idx = blockIdx.x * blockDim.x + threadIdx.x;
    if (idx >= BSN) return;
    int s = idx & (SN - 1);
    int b = idx / SN;
    int sm = (s + SN - 1) & (SN - 1), sp = (s + 1) & (SN - 1);
    const float* xm = x + ((size_t)b * SN + sm) * DIMN;
    const float* xc = x + ((size_t)b * SN + s)  * DIMN;
    const float* xp = x + ((size_t)b * SN + sp) * DIMN;
    float acc = bias[0];
    for (int d = 0; d < DIMN; ++d) {
        acc += w[d * 3 + 0] * xm[d] + w[d * 3 + 1] * xc[d] + w[d * 3 + 2] * xp[d];
    }
    out[idx] = acc;
}

// ============================================================================
// LayerNorm over last dim (DIM=256). One block (256 threads) per row.
// Writes fp32 and/or bf16 outputs (either may be null).
// ============================================================================
__global__ void layernorm_kernel(const float* __restrict__ in,
                                 const float* __restrict__ g,
                                 const float* __restrict__ b,
                                 float* __restrict__ outf,
                                 __bf16* __restrict__ outb)
{
    const int row = blockIdx.x;
    const int d = threadIdx.x;
    __shared__ float red[DIMN];
    float v = in[(size_t)row * DIMN + d];
    red[d] = v; __syncthreads();
    for (int o = DIMN / 2; o > 0; o >>= 1) {
        if (d < o) red[d] += red[d + o];
        __syncthreads();
    }
    float mean = red[0] * (1.0f / DIMN);
    __syncthreads();
    float diff = v - mean;
    red[d] = diff * diff; __syncthreads();
    for (int o = DIMN / 2; o > 0; o >>= 1) {
        if (d < o) red[d] += red[d + o];
        __syncthreads();
    }
    float var = red[0] * (1.0f / DIMN);
    float r = diff * rsqrtf(var + 1e-5f) * g[d] + b[d];
    if (outf) outf[(size_t)row * DIMN + d] = r;
    if (outb) outb[(size_t)row * DIMN + d] = (__bf16)r;
}

// ============================================================================
// Depthwise causal conv (K=4) + SiLU on xi (first DI cols of xz, stride 2*DI).
// flip=1 processes the sequence reversed (output stored in scan order).
// Writes fp32 u (for scan) and bf16 u (GEMM A operand).
// ============================================================================
__global__ void dwconv_silu_kernel(const float* __restrict__ xz,
                                   const float* __restrict__ cw,  // DI x K
                                   const float* __restrict__ cb,  // DI
                                   float* __restrict__ u,
                                   __bf16* __restrict__ ub, int flip)
{
    int idx = blockIdx.x * blockDim.x + threadIdx.x;
    if (idx >= BSN * DIN) return;
    int d = idx & (DIN - 1);
    int row = idx / DIN;
    int s = row & (SN - 1);
    int b = row / SN;
    float acc = cb[d];
#pragma unroll
    for (int t = 0; t < KCN; ++t) {
        int ss = s - (KCN - 1) + t;
        if (ss >= 0) {
            int src = flip ? (SN - 1 - ss) : ss;
            acc += cw[d * KCN + t] * xz[((size_t)b * SN + src) * (2 * DIN) + d];
        }
    }
    float r = siluf(acc);
    u[idx]  = r;
    ub[idx] = (__bf16)r;
}

// ============================================================================
// SSM scan. One wave32 handles 2 channels; lanes 0-15 = states of channel d0,
// lanes 16-31 = states of channel d1. Sequential over S; h kept in registers.
// ============================================================================
__global__ void ssm_scan_kernel(const float* __restrict__ u,    // BS x DI
                                const float* __restrict__ dt,   // BS x DI
                                const float* __restrict__ xdt,  // BS x XDTP
                                const float* __restrict__ A_log,// DI x NS
                                const float* __restrict__ Dp,   // DI
                                float* __restrict__ y,          // BS x DI
                                int flip)
{
    const int wave = threadIdx.x >> 5;
    const int lane = threadIdx.x & 31;
    const int half = lane >> 4;
    const int n = lane & 15;
    const int d = blockIdx.x * 16 + wave * 2 + half;   // channel
    const int b = blockIdx.y;
    const float Acoef = -__expf(A_log[(size_t)d * NSN + n]);
    const float dp = Dp[d];
    float h = 0.0f;
    for (int s = 0; s < SN; ++s) {
        size_t row = (size_t)b * SN + s;
        float dtv = dt[row * DIN + d];
        float uv  = u[row * DIN + d];
        float Bn  = xdt[row * XDTP + DTRN + n];
        float Cn  = xdt[row * XDTP + DTRN + NSN + n];
        h = __expf(dtv * Acoef) * h + dtv * uv * Bn;
        float p = h * Cn;
        p += __shfl_xor(p, 1, 16);
        p += __shfl_xor(p, 2, 16);
        p += __shfl_xor(p, 4, 16);
        p += __shfl_xor(p, 8, 16);
        if (n == 0) {
            int sout = flip ? (SN - 1 - s) : s;
            y[((size_t)b * SN + sout) * DIN + d] = p + dp * uv;
        }
    }
}

// ============================================================================
// Gate: g = (yf + yb) * silu(z)  (bf16 out: feeds W_out GEMM only)
// ============================================================================
__global__ void gate_kernel(const float* __restrict__ yf,
                            const float* __restrict__ yb,
                            const float* __restrict__ xz,
                            __bf16* __restrict__ g)
{
    int idx = blockIdx.x * blockDim.x + threadIdx.x;
    if (idx >= BSN * DIN) return;
    int d = idx & (DIN - 1);
    int row = idx / DIN;
    float z = xz[(size_t)row * (2 * DIN) + DIN + d];
    g[idx] = (__bf16)((yf[idx] + yb[idx]) * siluf(z));
}

// AdaLN modulate: out = silu(h*(scale+1)+shift); fp32 (for LN2) + bf16 (GEMM A)
__global__ void modulate_kernel(const float* __restrict__ h,
                                const float* __restrict__ cmod,
                                float* __restrict__ outf,
                                __bf16* __restrict__ outb, int per_pos)
{
    int idx = blockIdx.x * blockDim.x + threadIdx.x;
    if (idx >= BSN * DIMN) return;
    int d = idx & (DIMN - 1);
    int row = idx / DIMN;
    int b = row / SN;
    const float* cp = per_pos ? (cmod + (size_t)row * 2 * DIMN)
                              : (cmod + (size_t)b * 2 * DIMN);
    float sc = cp[d], sh = cp[DIMN + d];
    float r = siluf(h[idx] * (sc + 1.0f) + sh);
    outf[idx] = r;
    outb[idx] = (__bf16)r;
}

__global__ void add3_kernel(const float* __restrict__ a, const float* __restrict__ b,
                            const float* __restrict__ c, float* __restrict__ out, int n)
{
    int idx = blockIdx.x * blockDim.x + threadIdx.x;
    if (idx < n) out[idx] = a[idx] + b[idx] + c[idx];
}

__global__ void silu_bf16_kernel(const float* __restrict__ in,
                                 __bf16* __restrict__ out, int n)
{
    int idx = blockIdx.x * blockDim.x + threadIdx.x;
    if (idx < n) out[idx] = (__bf16)siluf(in[idx]);
}

// ============================================================================
// Host orchestration
// ============================================================================
struct StackPtrs {
    const float *mlp_w, *mlp_b, *n1_g, *n1_b, *n2_g, *n2_b;
    const float *W_in, *conv_w, *conv_b, *W_xdt, *W_dt, *b_dt, *A_log, *Dp, *W_out;
};

extern "C" void kernel_launch(void* const* d_in, const int* in_sizes, int n_in,
                              void* d_out, int out_size, void* d_ws, size_t ws_size,
                              hipStream_t stream)
{
    (void)in_sizes; (void)n_in; (void)out_size; (void)ws_size;

    const float* noised  = (const float*)d_in[0];
    const float* times   = (const float*)d_in[1];
    const float* cond    = (const float*)d_in[2];
    const float* time_w1 = (const float*)d_in[3];
    const float* time_b1 = (const float*)d_in[4];
    const float* time_w2 = (const float*)d_in[5];
    const float* time_b2 = (const float*)d_in[6];
    const float* icw     = (const float*)d_in[7];
    const float* icb     = (const float*)d_in[8];
    const float* fcw     = (const float*)d_in[9];
    const float* fcb     = (const float*)d_in[10];
    const float* down_w  = (const float*)d_in[11];
    const float* down_b  = (const float*)d_in[12];

    auto get_stack = [&](int base) {
        StackPtrs s;
        s.mlp_w  = (const float*)d_in[base + 0];
        s.mlp_b  = (const float*)d_in[base + 1];
        s.n1_g   = (const float*)d_in[base + 2];
        s.n1_b   = (const float*)d_in[base + 3];
        s.n2_g   = (const float*)d_in[base + 4];
        s.n2_b   = (const float*)d_in[base + 5];
        s.W_in   = (const float*)d_in[base + 6];
        s.conv_w = (const float*)d_in[base + 7];
        s.conv_b = (const float*)d_in[base + 8];
        s.W_xdt  = (const float*)d_in[base + 9];
        s.W_dt   = (const float*)d_in[base + 10];
        s.b_dt   = (const float*)d_in[base + 11];
        s.A_log  = (const float*)d_in[base + 12];
        s.Dp     = (const float*)d_in[base + 13];
        s.W_out  = (const float*)d_in[base + 14];
        return s;
    };
    StackPtrs enc = get_stack(13);
    StackPtrs dec = get_stack(28);

    // ---- workspace carving: fp32 activations ----
    float* wsf = (float*)d_ws;
    size_t off = 0;
    auto carve = [&](size_t nelem) { float* p = wsf + off; off += nelem; return p; };
    float* t_emb    = carve((size_t)BN * CDN);
    float* enc_cmod = carve((size_t)LN * BN * 2 * DIMN);
    float* cbuf     = carve((size_t)BSN * DIMN);
    float* xbuf     = carve((size_t)BSN * DIMN);
    float* cmod     = carve((size_t)BSN * 2 * DIMN);
    float* xzb      = carve((size_t)BSN * 2 * DIN);
    float* ubuf[2]  = { carve((size_t)BSN * DIN), carve((size_t)BSN * DIN) };
    float* xdtb[2]  = { carve((size_t)BSN * XDTP), carve((size_t)BSN * XDTP) };
    float* dtb[2]   = { carve((size_t)BSN * DIN), carve((size_t)BSN * DIN) };
    float* yfb      = carve((size_t)BSN * DIN);
    float* ybb      = carve((size_t)BSN * DIN);
    float* hbuf     = carve((size_t)BSN * DIMN);
    float* xnb      = carve((size_t)BSN * DIMN);
    float* h2b      = carve((size_t)BSN * DIMN);
    off = (off + 15) & ~(size_t)15;      // 16-float alignment for bf16 region

    // ---- workspace carving: bf16 (repacked weights Wt[N][Kp] + activations) ----
    __bf16* wsb = (__bf16*)(wsf + off);
    size_t boff = 0;
    auto bcarve = [&](size_t nelem) { __bf16* p = wsb + boff; boff += nelem; return p; };
    __bf16* down_wt  = bcarve((size_t)DIMN * CDN);                 // N=256, Kp=512
    __bf16* decmlp_t = bcarve((size_t)LN * (2 * DIMN) * DIMN);     // N=512, Kp=256 per layer
    __bf16* Wi_t[2]  = { bcarve((size_t)8 * (2*DIN) * DIMN), bcarve((size_t)8 * (2*DIN) * DIMN) }; // N=1024,Kp=256
    __bf16* Wx_t[2]  = { bcarve((size_t)8 * XDTN * DIN),     bcarve((size_t)8 * XDTN * DIN) };     // N=48, Kp=512
    __bf16* Wd_t[2]  = { bcarve((size_t)8 * DIN * 32),       bcarve((size_t)8 * DIN * 32) };       // N=512,Kp=32
    __bf16* Wo_t[2]  = { bcarve((size_t)8 * DIMN * DIN),     bcarve((size_t)8 * DIMN * DIN) };     // N=256,Kp=512
    // bf16 activation shadows (GEMM A operands)
    __bf16* cond_bf  = bcarve((size_t)BSN * CDN);
    __bf16* siluc_bf = bcarve((size_t)BSN * DIMN);
    __bf16* ln1_bf   = bcarve((size_t)BSN * DIMN);
    __bf16* h_bf     = bcarve((size_t)BSN * DIMN);
    __bf16* u_bf[2]  = { bcarve((size_t)BSN * DIN), bcarve((size_t)BSN * DIN) };
    __bf16* xdt_bf[2]= { bcarve((size_t)BSN * XDTP), bcarve((size_t)BSN * XDTP) };
    __bf16* g_bf     = bcarve((size_t)BSN * DIN);

    const int TPB = 256;
    auto nblk = [&](int n) { return (n + TPB - 1) / TPB; };

    auto repack = [&](const float* W, __bf16* Wt, int K, int N, int Kp) {
        repack_weight_kernel<<<nblk(N * Kp), TPB, 0, stream>>>(W, Wt, K, N, Kp);
    };

    // ---- repack all GEMM weights (bf16, transposed, K zero-padded) ----
    repack(down_w, down_wt, CDN, DIMN, CDN);
    for (int i = 0; i < LN; ++i)
        repack(dec.mlp_w + (size_t)i * DIMN * 2 * DIMN,
               decmlp_t + (size_t)i * (2 * DIMN) * DIMN, DIMN, 2 * DIMN, DIMN);
    const StackPtrs* stacks[2] = { &enc, &dec };
    for (int sidx = 0; sidx < 2; ++sidx) {
        const StackPtrs& st = *stacks[sidx];
        for (int od = 0; od < 8; ++od) {
            repack(st.W_in  + (size_t)od * DIMN * 2 * DIN,
                   Wi_t[sidx] + (size_t)od * (2 * DIN) * DIMN, DIMN, 2 * DIN, DIMN);
            repack(st.W_xdt + (size_t)od * DIN * XDTN,
                   Wx_t[sidx] + (size_t)od * XDTN * DIN, DIN, XDTN, DIN);
            repack(st.W_dt  + (size_t)od * DTRN * DIN,
                   Wd_t[sidx] + (size_t)od * DIN * 32, DTRN, DIN, 32);
            repack(st.W_out + (size_t)od * DIN * DIMN,
                   Wo_t[sidx] + (size_t)od * DIMN * DIN, DIN, DIMN, DIN);
        }
    }

    auto gemm = [&](const __bf16* A, int lda, const __bf16* Wt, int Kp,
                    const float* bias, float* C, int ldc,
                    __bf16* Cb, int ldcb, int N, int act) {
        dim3 blk(32, 8);
        dim3 grd((N / 16 + 7) / 8, BSN / (16 * MT));
        wmma_gemm_kernel<<<grd, blk, 0, stream>>>(A, lda, Wt, Kp, bias, C, ldc,
                                                  Cb, ldcb, N, act);
    };

    auto hydra = [&](const __bf16* xin_bf, float* yout, const StackPtrs& st, int sidx,
                     int i, int j, int act_out) {
        size_t od = (size_t)(i * 2 + j);
        const float* cw  = st.conv_w + od * DIN * KCN;
        const float* cb  = st.conv_b + od * DIN;
        const float* bd  = st.b_dt   + od * DIN;
        const float* Al  = st.A_log  + od * DIN * NSN;
        const float* Dpp = st.Dp     + od * DIN;

        // xz = xin @ W_in        (N=1024, K=256)
        gemm(xin_bf, DIMN, Wi_t[sidx] + od * (2 * DIN) * DIMN, DIMN,
             nullptr, xzb, 2 * DIN, nullptr, 0, 2 * DIN, 0);
        for (int dir = 0; dir < 2; ++dir) {
            dwconv_silu_kernel<<<nblk(BSN * DIN), TPB, 0, stream>>>(
                xzb, cw, cb, ubuf[dir], u_bf[dir], dir);
            // xdt = u @ W_xdt    (N=48, K=512) -> fp32 (scan) + bf16 shadow (dt GEMM)
            gemm(u_bf[dir], DIN, Wx_t[sidx] + od * XDTN * DIN, DIN,
                 nullptr, xdtb[dir], XDTP, xdt_bf[dir], XDTP, XDTN, 0);
            // dt = softplus(dtr @ W_dt + b_dt)   (N=512, Kp=32, A = xdt_bf stride 64)
            gemm(xdt_bf[dir], XDTP, Wd_t[sidx] + od * DIN * 32, 32,
                 bd, dtb[dir], DIN, nullptr, 0, DIN, 2);
            ssm_scan_kernel<<<dim3(DIN / 16, BN), TPB, 0, stream>>>(
                ubuf[dir], dtb[dir], xdtb[dir], Al, Dpp, dir ? ybb : yfb, dir);
        }
        gate_kernel<<<nblk(BSN * DIN), TPB, 0, stream>>>(yfb, ybb, xzb, g_bf);
        // out = g @ W_out        (N=256, K=512)
        gemm(g_bf, DIN, Wo_t[sidx] + od * DIMN * DIN, DIN,
             nullptr, yout, DIMN, nullptr, 0, DIMN, act_out);
    };

    auto hydra_layer = [&](float* state, const StackPtrs& st, int sidx, int i,
                           const float* cmod_p, int per_pos) {
        layernorm_kernel<<<BSN, DIMN, 0, stream>>>(state, st.n1_g + (size_t)i * DIMN,
                                                   st.n1_b + (size_t)i * DIMN,
                                                   nullptr, ln1_bf);
        hydra(ln1_bf, h2b, st, sidx, i, 0, 0);
        modulate_kernel<<<nblk(BSN * DIMN), TPB, 0, stream>>>(h2b, cmod_p, hbuf, h_bf, per_pos);
        layernorm_kernel<<<BSN, DIMN, 0, stream>>>(hbuf, st.n2_g + (size_t)i * DIMN,
                                                   st.n2_b + (size_t)i * DIMN,
                                                   xnb, (__bf16*)nullptr);
        hydra(h_bf, h2b, st, sidx, i, 1, 1);  // silu fused into W_out epilogue
        add3_kernel<<<nblk(BSN * DIMN), TPB, 0, stream>>>(h2b, xnb, state, state, BSN * DIMN);
    };

    // ---- forward pass ----
    time_embed_kernel<<<BN, 256, 0, stream>>>(times, time_w1, time_b1, time_w2, time_b2, t_emb);
    enc_cmod_kernel<<<dim3(LN, BN), 256, 0, stream>>>(t_emb, enc.mlp_w, enc.mlp_b, enc_cmod);
    init_conv_kernel<<<nblk(BSN * DIMN), TPB, 0, stream>>>(noised, icw, icb, xbuf);
    // c = cond @ down_w + down_b   (N=256, K=512); cond cast to bf16 once
    cast_bf16_kernel<<<nblk(BSN * CDN), TPB, 0, stream>>>(cond, cond_bf, BSN * CDN);
    gemm(cond_bf, CDN, down_wt, CDN, down_b, cbuf, DIMN, nullptr, 0, DIMN, 0);

    for (int i = 0; i < LN; ++i) {
        hydra_layer(cbuf, enc, 0, i, enc_cmod + (size_t)i * BN * 2 * DIMN, /*per_pos=*/0);
    }

    silu_bf16_kernel<<<nblk(BSN * DIMN), TPB, 0, stream>>>(cbuf, siluc_bf, BSN * DIMN);

    for (int i = 0; i < LN; ++i) {
        // cmod = silu(c) @ dec.mlp_w[i] + mlp_b[i]   (N=512, K=256)
        gemm(siluc_bf, DIMN, decmlp_t + (size_t)i * (2 * DIMN) * DIMN, DIMN,
             dec.mlp_b + (size_t)i * 2 * DIMN, cmod, 2 * DIMN, nullptr, 0, 2 * DIMN, 0);
        hydra_layer(xbuf, dec, 1, i, cmod, /*per_pos=*/1);
    }

    final_conv_kernel<<<nblk(BSN), TPB, 0, stream>>>(xbuf, fcw, fcb, (float*)d_out);
}
